// GraphNCA_76785425317998
// MI455X (gfx1250) — matrix operand order repaired and checked by reading, not compiled, not run on hardware
//
#include <hip/hip_runtime.h>

typedef _Float16 f16;
typedef _Float16 v16h __attribute__((ext_vector_type(16)));
typedef _Float16 v8h  __attribute__((ext_vector_type(8)));
typedef float    v8f  __attribute__((ext_vector_type(8)));

#define B_   16
#define C_   32
#define P_   8
#define N_   64
#define IMG  1024      // B_*N_
#define PIX  256       // 16*16
#define D_   64
#define HID_ 128
#define KIT  8

// Use CDNA5 GLOBAL_LOAD_ASYNC_TO_LDS_B128 (ASYNCcnt) for conv staging.
#define USE_ASYNC_LDS 1

enum { ACT_NONE = 0, ACT_ELU = 1, ACT_SIG = 2, ACT_TANH = 3, ACT_RELU = 4 };

template <int ACT>
__device__ __forceinline__ float act_apply(float v) {
  if (ACT == ACT_ELU)  return v > 0.f ? v : (expf(v) - 1.f);
  if (ACT == ACT_SIG)  return 1.f / (1.f + expf(-v));
  if (ACT == ACT_TANH) return tanhf(v);
  if (ACT == ACT_RELU) return v > 0.f ? v : 0.f;
  return v;
}

// ============================ weight pre-shuffle (fragment-ready layouts) ===
// conv A-fragment: K reordered as k = (ky*3+kx)*cin + ci (tap-major).
// out[((mt*nk + ks)*32 + lane)*16 + j] : lane's 16 halfs contiguous.
__global__ void k_shuf_convW(const float* __restrict__ W, f16* __restrict__ out,
                             int cin, int cout) {
  int nk = cin * 9 / 32;
  int total = (cout / 16) * nk * 512;
  int idx = blockIdx.x * blockDim.x + threadIdx.x;
  if (idx >= total) return;
  int j = idx & 15;
  int lane = (idx >> 4) & 31;
  int rest = idx >> 9;  // mt*nk + ks
  int ks = rest % nk, mt = rest / nk;
  int m = mt * 16 + (lane & 15);
  int kk = ks * 32 + ((lane >> 4) ? 16 : 0) + j;
  int ci = kk % cin, tap = kk / cin;
  int ky = tap / 3, kx = tap - ky * 3;
  out[idx] = (f16)W[((m * cin + ci) * 3 + ky) * 3 + kx];
}

// GEMM B-fragment (W is [N,K] row-major; B[k,n] = W[n,k]).
__global__ void k_shuf_gemmW(const float* __restrict__ W, f16* __restrict__ out,
                             int N, int K) {
  int nk = K / 32;
  int total = (N / 16) * nk * 512;
  int idx = blockIdx.x * blockDim.x + threadIdx.x;
  if (idx >= total) return;
  int j = idx & 15;
  int lane = (idx >> 4) & 31;
  int rest = idx >> 9;  // nt*nk + ks
  int ks = rest % nk, nt = rest / nk;
  int ncol = nt * 16 + (lane & 15);
  int kk = ks * 32 + j + ((j >= 8) ? 8 : 0) + ((lane >> 4) ? 8 : 0);
  out[idx] = (f16)W[ncol * K + kk];
}

// ================================================================ utilities
// seed (B,C,128,128) -> X pixel-major (img, pix, c) with clip(-10,10)
__global__ void k_init(const float* __restrict__ seed, float* __restrict__ X) {
  int idx = blockIdx.x * blockDim.x + threadIdx.x;
  if (idx >= IMG * PIX * C_) return;
  int c = idx & 31, pix = (idx >> 5) & 255, img = idx >> 13;
  int b = img >> 6, n = img & 63;
  int p = n >> 3, q = n & 7;
  int y = pix >> 4, x = pix & 15;
  float v = seed[(((size_t)b * C_ + c) * 128 + (p * 16 + y)) * 128 + (q * 16 + x)];
  X[idx] = fminf(10.f, fmaxf(-10.f, v));
}

// X f32 (img,pix,c) -> Xh f16 same layout, plus spatial mean per (img,c).
__global__ void k_pack(const float* __restrict__ X, f16* __restrict__ Xh,
                       f16* __restrict__ xmean) {
  __shared__ float red[256];
  int img = blockIdx.x, tid = threadIdx.x;
  const float* Xi = X + (size_t)img * PIX * C_;
  f16* Xo = Xh + (size_t)img * PIX * C_;
  float part = 0.f;  // partial sum for channel c = tid & 31
  for (int it = 0; it < 32; ++it) {
    int idx = it * 256 + tid;  // c = idx&31 == tid&31 (256 % 32 == 0)
    float v = Xi[idx];
    Xo[idx] = (f16)v;
    part += v;
  }
  red[tid] = part;
  __syncthreads();
  if (tid < 32) {
    float s = 0.f;
    for (int g = 0; g < 8; ++g) s += red[tid + 32 * g];
    xmean[img * C_ + tid] = (f16)(s * (1.f / 256.f));
  }
}

// ------------------------------------------------- 3x3 conv as implicit GEMM
// LDS: zero-padded 18x18 x CIN, channel-minor -> B-fragments are 32B
// contiguous (tap-major K order, runs never cross a CIN boundary).
template <int CIN, int CSPLIT, int COUT, int ACT>
__global__ void __launch_bounds__(128) k_conv3x3_wmma(
    const f16* __restrict__ inA, const f16* __restrict__ inB,
    const v16h* __restrict__ Wf, const float* __restrict__ bias,
    f16* __restrict__ out) {
  constexpr int NK = CIN * 9 / 32;
  constexpr int MT = COUT / 16;
  constexpr int CB = CIN - CSPLIT;
  __shared__ alignas(32) f16 smem[324 * CIN];
  int img = blockIdx.x, tid = threadIdx.x;

#if USE_ASYNC_LDS
  // 1) zero the padding border with wide LDS stores (disjoint from interior)
  for (int idx = tid; idx < 324 * (CIN / 8); idx += 128) {
    int s = idx / (CIN / 8);
    int py = s / 18, px = s - py * 18;
    if (py == 0 || py == 17 || px == 0 || px == 17) {
      v8h z = {};
      *(v8h*)&smem[(size_t)idx * 8] = z;
    }
  }
  // 2) async-copy the 16x16 interior straight into LDS (ASYNCcnt tracked).
  //    GVS form: vdst = LDS byte offset, vaddr = 32-bit offset, saddr = base.
  unsigned ldsbase = (unsigned)(size_t)(&smem[0]);  // LDS aperture: low 32 bits
  for (int idx = tid; idx < 256 * (CSPLIT / 8); idx += 128) {
    int pixel = idx / (CSPLIT / 8);
    int c = (idx & (CSPLIT / 8 - 1)) * 8;
    int y = pixel >> 4, x = pixel & 15;
    unsigned lds = ldsbase + (unsigned)((((y + 1) * 18 + (x + 1)) * CIN + c) << 1);
    int voff = (int)((((unsigned)img * PIX + pixel) * CSPLIT + c) << 1);
    asm volatile("global_load_async_to_lds_b128 %0, %1, %2"
                 :: "v"(lds), "v"(voff), "s"(inA) : "memory");
  }
  if (CB > 0) {
    for (int idx = tid; idx < 256 * (CB / 8); idx += 128) {
      int pixel = idx / (CB / 8);
      int c = (idx & (CB / 8 - 1)) * 8;
      int y = pixel >> 4, x = pixel & 15;
      unsigned lds =
          ldsbase + (unsigned)((((y + 1) * 18 + (x + 1)) * CIN + CSPLIT + c) << 1);
      int voff = (int)((((unsigned)img * PIX + pixel) * CB + c) << 1);
      asm volatile("global_load_async_to_lds_b128 %0, %1, %2"
                   :: "v"(lds), "v"(voff), "s"(inB) : "memory");
    }
  }
  asm volatile("s_wait_asynccnt 0" ::: "memory");
  __syncthreads();
#else
  for (int idx = tid; idx < 324 * CIN; idx += 128) {
    int c = idx & (CIN - 1);
    int s = idx / CIN;
    int py = s / 18, px = s - py * 18;
    int y = py - 1, x = px - 1;
    f16 v = (f16)0.f;
    if (y >= 0 && y < 16 && x >= 0 && x < 16) {
      int pix = y * 16 + x;
      v = (c < CSPLIT) ? inA[((size_t)img * PIX + pix) * CSPLIT + c]
                       : inB[((size_t)img * PIX + pix) * CB + (c - CSPLIT)];
    }
    smem[idx] = v;
  }
  __syncthreads();
#endif

  int wave = tid >> 5, lane = tid & 31, l15 = lane & 15, hi = lane >> 4;
  for (int nt = wave; nt < 16; nt += 4) {
    int pix = nt * 16 + l15, yy = pix >> 4, xx = pix & 15;
    v8f acc[MT];
#pragma unroll
    for (int mt = 0; mt < MT; ++mt)
#pragma unroll
      for (int i = 0; i < 8; ++i) acc[mt][i] = 0.f;
    for (int ks = 0; ks < NK; ++ks) {
      int kb = ks * 32 + hi * 16;
      int tap = kb / CIN, cb = kb - tap * CIN;
      int ky = tap / 3, kx = tap - ky * 3;
      v16h b = *(const v16h*)&smem[((yy + ky) * 18 + (xx + kx)) * CIN + cb];
#pragma unroll
      for (int mt = 0; mt < MT; ++mt) {
        v16h a = Wf[(mt * NK + ks) * 32 + lane];
        acc[mt] = __builtin_amdgcn_wmma_f32_16x16x32_f16(
            false, a, false, b, (short)0, acc[mt], false, false);
      }
    }
#pragma unroll
    for (int mt = 0; mt < MT; ++mt) {
      v8f bv = *(const v8f*)&bias[mt * 16 + hi * 8];  // contiguous 32B
      v8h st;
#pragma unroll
      for (int i = 0; i < 8; ++i) st[i] = (f16)act_apply<ACT>(acc[mt][i] + bv[i]);
      *(v8h*)&out[((size_t)img * PIX + pix) * COUT + mt * 16 + hi * 8] = st;
    }
  }
}

// ---------------------------------------------------------- generic GEMM
// out[M,N] = act(A[M,K] @ W^T + bias[N]); A row-major, W pre-shuffled.
template <int ACT>
__global__ void __launch_bounds__(128) k_gemm_wmma(
    const f16* __restrict__ A, const v16h* __restrict__ Wf,
    const float* __restrict__ bias, f16* __restrict__ out,
    int M, int N, int K) {
  int tid = threadIdx.x, lane = tid & 31, l15 = lane & 15, hi = lane >> 4;
  int gwave = (blockIdx.x * blockDim.x + tid) >> 5;
  int nwaves = (gridDim.x * blockDim.x) >> 5;
  int ntile = N >> 4, nk = K >> 5;
  int totalTiles = (M >> 4) * ntile;
  for (int t = gwave; t < totalTiles; t += nwaves) {
    int mt = t / ntile, nt = t - mt * ntile;
    v8f acc;
#pragma unroll
    for (int i = 0; i < 8; ++i) acc[i] = 0.f;
    int mrow = mt * 16 + l15, ncol = nt * 16 + l15;
    const f16* Arow = A + (size_t)mrow * K + hi * 8;
    for (int ks = 0; ks < nk; ++ks) {
      v8h lo = *(const v8h*)(Arow + ks * 32);
      v8h h8 = *(const v8h*)(Arow + ks * 32 + 16);
      v16h a = __builtin_shufflevector(lo, h8, 0, 1, 2, 3, 4, 5, 6, 7, 8, 9,
                                       10, 11, 12, 13, 14, 15);
      v16h b = Wf[(nt * nk + ks) * 32 + lane];
      acc = __builtin_amdgcn_wmma_f32_16x16x32_f16(false, a, false, b,
                                                   (short)0, acc, false, false);
    }
#pragma unroll
    for (int i = 0; i < 8; ++i) {
      int m = mt * 16 + hi * 8 + i;
      float v = acc[i] + bias[ncol];
      out[(size_t)m * N + ncol] = (f16)act_apply<ACT>(v);
    }
  }
}

// -------------------------------------------- mix 1x1 conv (pixel-major GEMM)
// rows = IMG*PIX; K=64 from concat(Xloc,Xg) (both pixel-major, 32 ch);
// writes f32 state pixel-major.
__global__ void __launch_bounds__(128) k_mix_wmma(
    const f16* __restrict__ Xloc, const f16* __restrict__ Xg,
    const v16h* __restrict__ Wf, const float* __restrict__ bias,
    float* __restrict__ Xout) {
  int tid = threadIdx.x, lane = tid & 31, l15 = lane & 15, hi = lane >> 4;
  int gwave = (blockIdx.x * blockDim.x + tid) >> 5;
  int nwaves = (gridDim.x * blockDim.x) >> 5;
  int totalTiles = ((IMG * PIX) >> 4) * 2;
  for (int t = gwave; t < totalTiles; t += nwaves) {
    int mt = t >> 1, nt = t & 1;
    v8f acc;
#pragma unroll
    for (int i = 0; i < 8; ++i) acc[i] = 0.f;
    int mrow = mt * 16 + l15;
    int img = mrow >> 8, pix = mrow & 255;
    int ncol = nt * 16 + l15;
    const f16* baseL = Xloc + ((size_t)img * PIX + pix) * C_;
    const f16* baseG = Xg + ((size_t)img * PIX + pix) * C_;
#pragma unroll
    for (int ks = 0; ks < 2; ++ks) {
      int kb0 = ks * 32 + hi * 8;       // channels for a[0..7]
      int kb1 = ks * 32 + 16 + hi * 8;  // channels for a[8..15]
      v8h lo = *(const v8h*)((kb0 < C_) ? (baseL + kb0) : (baseG + kb0 - C_));
      v8h h8 = *(const v8h*)((kb1 < C_) ? (baseL + kb1) : (baseG + kb1 - C_));
      v16h a = __builtin_shufflevector(lo, h8, 0, 1, 2, 3, 4, 5, 6, 7, 8, 9,
                                       10, 11, 12, 13, 14, 15);
      v16h b = Wf[(nt * 2 + ks) * 32 + lane];
      acc = __builtin_amdgcn_wmma_f32_16x16x32_f16(false, a, false, b,
                                                   (short)0, acc, false, false);
    }
#pragma unroll
    for (int i = 0; i < 8; ++i) {
      int m = mt * 16 + hi * 8 + i;
      int oimg = m >> 8, opix = m & 255;
      Xout[((size_t)oimg * PIX + opix) * C_ + ncol] = acc[i] + bias[ncol];
    }
  }
}

// ------------------------------------------------------------- edge pipeline
__global__ void k_gather_ein(const f16* __restrict__ qv, const f16* __restrict__ kv,
                             const int* __restrict__ ei, f16* __restrict__ ein,
                             int E) {
  int idx = blockIdx.x * blockDim.x + threadIdx.x;
  if (idx >= B_ * E * HID_) return;
  int k = idx & 127;
  int t = idx >> 7;  // b*E + e
  int e = t % E, b = t / E;
  int s = ei[e * 2 + 0], d = ei[e * 2 + 1];
  ein[idx] = (k < D_) ? qv[(b * N_ + s) * D_ + k] : kv[(b * N_ + d) * D_ + (k - D_)];
}

__global__ void k_eval(const f16* __restrict__ h2, const float* __restrict__ w3,
                       const float* __restrict__ b3, float* __restrict__ ev,
                       int rows) {
  int r = blockIdx.x * blockDim.x + threadIdx.x;
  if (r >= rows) return;
  float s = 0.f;
  for (int k = 0; k < HID_; ++k) s += (float)h2[(size_t)r * HID_ + k] * w3[k];
  ev[r] = s + b3[0];
}

// S[c,pix] = sum of gate_w taps valid at this border position
__global__ void k_gateS(const float* __restrict__ gw, float* __restrict__ S) {
  int idx = blockIdx.x * blockDim.x + threadIdx.x;
  if (idx >= C_ * PIX) return;
  int pix = idx & 255, c = idx >> 8;
  int y = pix >> 4, x = pix & 15;
  float s = 0.f;
  for (int ky = 0; ky < 3; ++ky)
    for (int kx = 0; kx < 3; ++kx) {
      int yy = y + ky - 1, xx = x + kx - 1;
      if (yy >= 0 && yy < 16 && xx >= 0 && xx < 16) s += gw[c * 9 + ky * 3 + kx];
    }
  S[idx] = s;
}

__global__ void k_slot_init(int* __restrict__ eslot, int* __restrict__ dslot) {
  int i = threadIdx.x;
  if (i < N_ * 4) { eslot[i] = -1; dslot[i] = -1; }
}
__global__ void k_slot_build(const int* __restrict__ ei, int* __restrict__ eslot,
                             int* __restrict__ dslot, int E) {
  int e = blockIdx.x * blockDim.x + threadIdx.x;
  if (e >= E) return;
  int s = ei[e * 2], d = ei[e * 2 + 1];
  int diff = d - s;
  int slot = (diff == -P_) ? 0 : (diff == P_) ? 1 : (diff == -1) ? 2 : 3;
  eslot[s * 4 + slot] = e;
  dslot[s * 4 + slot] = d;
}

// agg[b,i,pix,c] = sum over edges (src=i,dst=j): sigmoid(e*S + gb) * M[b,j]
__global__ void k_agg(const float* __restrict__ ev, const int* __restrict__ eslot,
                      const int* __restrict__ dslot, const float* __restrict__ S,
                      const float* __restrict__ gate_b, const f16* __restrict__ MH,
                      f16* __restrict__ MaggH, int E) {
  __shared__ float sev[4];
  __shared__ int sdn[4];
  int img = blockIdx.x, tid = threadIdx.x;
  int b = img >> 6, node = img & 63;
  if (tid < 4) {
    int e = eslot[node * 4 + tid];
    sdn[tid] = (e >= 0) ? dslot[node * 4 + tid] : -1;
    sev[tid] = (e >= 0) ? ev[b * E + e] : 0.f;
  }
  __syncthreads();
  for (int it = 0; it < 32; ++it) {
    int idx = it * 256 + tid;  // (pix, c), c fastest -> coalesced
    int c = idx & 31, pix = idx >> 5;
    float sc = S[c * PIX + pix], gb = gate_b[c], acc = 0.f;
#pragma unroll
    for (int s = 0; s < 4; ++s) {
      int d = sdn[s];
      if (d >= 0) {
        float g = 1.f / (1.f + expf(-(sev[s] * sc + gb)));
        acc += g * (float)MH[(((size_t)b * N_ + d) * PIX + pix) * C_ + c];
      }
    }
    MaggH[(size_t)img * PIX * C_ + idx] = (f16)acc;
  }
}

// ------------------------------------------------------------- GRU elementwise
__global__ void k_rh(const f16* __restrict__ gates, const float* __restrict__ X,
                     f16* __restrict__ rh) {
  int idx = blockIdx.x * blockDim.x + threadIdx.x;
  if (idx >= IMG * PIX * C_) return;
  int c = idx & 31, rest = idx >> 5;  // img*PIX + pix
  float r = (float)gates[(size_t)rest * 2 * C_ + c];
  rh[idx] = (f16)(r * X[idx]);
}
__global__ void k_xgraph(const f16* __restrict__ gates, const float* __restrict__ X,
                         const f16* __restrict__ cand, f16* __restrict__ Xg) {
  int idx = blockIdx.x * blockDim.x + threadIdx.x;
  if (idx >= IMG * PIX * C_) return;
  int c = idx & 31, rest = idx >> 5;
  float z = (float)gates[(size_t)rest * 2 * C_ + C_ + c];
  Xg[idx] = (f16)((1.f - z) * X[idx] + z * (float)cand[idx]);
}

// ------------------------------------------------------------- final output
__global__ void k_final(const float* __restrict__ X, float* __restrict__ out) {
  int idx = blockIdx.x * blockDim.x + threadIdx.x;
  if (idx >= B_ * C_ * 128 * 128) return;
  int gx = idx & 127, gy = (idx >> 7) & 127;
  int c = (idx >> 14) & 31, b = idx >> 19;
  int p = gy >> 4, y = gy & 15, q = gx >> 4, x = gx & 15;
  size_t base = ((size_t)(b * N_ + p * P_ + q) * PIX + (y * 16 + x)) * C_;
  float v = X[base + c];
  if (c >= 1 && c <= 3) {
    float a0 = X[base];  // channel 0
    a0 = fminf(1.f, fmaxf(0.f, a0));
    v *= a0;
  }
  out[idx] = v;
}

// ============================================================== host driver
extern "C" void kernel_launch(void* const* d_in, const int* in_sizes, int n_in,
                              void* d_out, int out_size, void* d_ws, size_t ws_size,
                              hipStream_t stream) {
  const float* seed    = (const float*)d_in[0];
  const int*   ei      = (const int*)  d_in[1];
  const float* local_w = (const float*)d_in[2];
  const float* local_b = (const float*)d_in[3];
  const float* fq_w    = (const float*)d_in[4];
  const float* fq_b    = (const float*)d_in[5];
  const float* fk_w    = (const float*)d_in[6];
  const float* fk_b    = (const float*)d_in[7];
  const float* fm_w    = (const float*)d_in[8];
  const float* fm_b    = (const float*)d_in[9];
  const float* attn_w1 = (const float*)d_in[10];
  const float* attn_b1 = (const float*)d_in[11];
  const float* attn_w2 = (const float*)d_in[12];
  const float* attn_b2 = (const float*)d_in[13];
  const float* attn_w3 = (const float*)d_in[14];
  const float* attn_b3 = (const float*)d_in[15];
  const float* gate_w  = (const float*)d_in[16];
  const float* gate_b  = (const float*)d_in[17];
  const float* gru_gw  = (const float*)d_in[18];
  const float* gru_gb  = (const float*)d_in[19];
  const float* gru_cw  = (const float*)d_in[20];
  const float* gru_cb  = (const float*)d_in[21];
  const float* mix_w   = (const float*)d_in[22];
  const float* mix_b   = (const float*)d_in[23];

  const int E = in_sizes[1] / 2;  // 224

  char* base = (char*)d_ws;
  size_t off = 0;
  auto carve = [&](size_t bytes) -> char* {
    char* p = base + off;
    off += (bytes + 255) & ~(size_t)255;
    return p;
  };

  const size_t IC = (size_t)IMG * PIX * C_;
  float* Xa     = (float*)carve(IC * 4);
  float* Xb     = (float*)carve(IC * 4);
  f16* Xh       = (f16*)carve(IC * 2);
  f16* XlocH    = (f16*)carve(IC * 2);
  f16* MH       = (f16*)carve(IC * 2);
  f16* MaggH    = (f16*)carve(IC * 2);
  f16* gatesH   = (f16*)carve(IC * 4);  // 64 channels
  f16* rhH      = (f16*)carve(IC * 2);
  f16* candH    = (f16*)carve(IC * 2);
  f16* XgH      = (f16*)carve(IC * 2);
  f16* xmeanH   = (f16*)carve((size_t)IMG * C_ * 2);
  f16* qvH      = (f16*)carve((size_t)IMG * D_ * 2);
  f16* kvH      = (f16*)carve((size_t)IMG * D_ * 2);
  f16* einH     = (f16*)carve((size_t)B_ * E * HID_ * 2);
  f16* h1H      = (f16*)carve((size_t)B_ * E * HID_ * 2);
  f16* h2H      = (f16*)carve((size_t)B_ * E * HID_ * 2);
  float* evF    = (float*)carve((size_t)B_ * E * 4);
  float* Sb     = (float*)carve((size_t)C_ * PIX * 4);
  int* eslot    = (int*)carve((size_t)N_ * 4 * 4);
  int* dslot    = (int*)carve((size_t)N_ * 4 * 4);
  f16* localWf  = (f16*)carve(32 * 288 * 2);   // fragment-ready
  f16* fmWf     = (f16*)carve(32 * 288 * 2);
  f16* ggWf     = (f16*)carve(64 * 576 * 2);
  f16* gcWf     = (f16*)carve(32 * 576 * 2);
  f16* fqWf     = (f16*)carve(64 * 32 * 2);
  f16* fkWf     = (f16*)carve(64 * 32 * 2);
  f16* w1f      = (f16*)carve(128 * 128 * 2);
  f16* w2f      = (f16*)carve(128 * 128 * 2);
  f16* mixWf    = (f16*)carve(32 * 64 * 2);

  auto shufC = [&](const float* s, f16* d, int cin, int cout) {
    int n = (cout / 16) * (cin * 9 / 32) * 512;
    k_shuf_convW<<<(n + 255) / 256, 256, 0, stream>>>(s, d, cin, cout);
  };
  auto shufG = [&](const float* s, f16* d, int N, int K) {
    int n = (N / 16) * (K / 32) * 512;
    k_shuf_gemmW<<<(n + 255) / 256, 256, 0, stream>>>(s, d, N, K);
  };
  shufC(local_w, localWf, 32, 32);
  shufC(fm_w, fmWf, 32, 32);
  shufC(gru_gw, ggWf, 64, 64);
  shufC(gru_cw, gcWf, 64, 32);
  shufG(fq_w, fqWf, 64, 32);
  shufG(fk_w, fkWf, 64, 32);
  shufG(attn_w1, w1f, 128, 128);
  shufG(attn_w2, w2f, 128, 128);
  shufG(mix_w, mixWf, 32, 64);

  k_gateS<<<(C_ * PIX + 255) / 256, 256, 0, stream>>>(gate_w, Sb);
  k_slot_init<<<1, 256, 0, stream>>>(eslot, dslot);
  k_slot_build<<<1, 256, 0, stream>>>(ei, eslot, dslot, E);
  k_init<<<(int)((IC + 255) / 256), 256, 0, stream>>>(seed, Xa);

  const int EW = (int)((IC + 255) / 256);
  float* Xc = Xa;
  float* Xn = Xb;
  for (int t = 0; t < KIT; ++t) {
    k_pack<<<IMG, 256, 0, stream>>>(Xc, Xh, xmeanH);
    k_conv3x3_wmma<32, 32, 32, ACT_ELU><<<IMG, 128, 0, stream>>>(
        Xh, nullptr, (const v16h*)localWf, local_b, XlocH);
    k_conv3x3_wmma<32, 32, 32, ACT_NONE><<<IMG, 128, 0, stream>>>(
        Xh, nullptr, (const v16h*)fmWf, fm_b, MH);
    k_gemm_wmma<ACT_NONE><<<64, 128, 0, stream>>>(
        xmeanH, (const v16h*)fqWf, fq_b, qvH, IMG, D_, C_);
    k_gemm_wmma<ACT_NONE><<<64, 128, 0, stream>>>(
        xmeanH, (const v16h*)fkWf, fk_b, kvH, IMG, D_, C_);
    k_gather_ein<<<(B_ * E * HID_ + 255) / 256, 256, 0, stream>>>(qvH, kvH, ei,
                                                                  einH, E);
    k_gemm_wmma<ACT_RELU><<<448, 128, 0, stream>>>(
        einH, (const v16h*)w1f, attn_b1, h1H, B_ * E, HID_, HID_);
    k_gemm_wmma<ACT_RELU><<<448, 128, 0, stream>>>(
        h1H, (const v16h*)w2f, attn_b2, h2H, B_ * E, HID_, HID_);
    k_eval<<<(B_ * E + 255) / 256, 256, 0, stream>>>(h2H, attn_w3, attn_b3, evF,
                                                     B_ * E);
    k_agg<<<IMG, 256, 0, stream>>>(evF, eslot, dslot, Sb, gate_b, MH, MaggH, E);
    k_conv3x3_wmma<64, 32, 64, ACT_SIG><<<IMG, 128, 0, stream>>>(
        MaggH, Xh, (const v16h*)ggWf, gru_gb, gatesH);
    k_rh<<<EW, 256, 0, stream>>>(gatesH, Xc, rhH);
    k_conv3x3_wmma<64, 32, 32, ACT_TANH><<<IMG, 128, 0, stream>>>(
        MaggH, rhH, (const v16h*)gcWf, gru_cb, candH);
    k_xgraph<<<EW, 256, 0, stream>>>(gatesH, Xc, candH, XgH);
    k_mix_wmma<<<8192, 128, 0, stream>>>(XlocH, XgH, (const v16h*)mixWf, mix_b, Xn);
    float* tmp = Xc; Xc = Xn; Xn = tmp;
  }
  k_final<<<(B_ * C_ * 128 * 128 + 255) / 256, 256, 0, stream>>>(Xc, (float*)d_out);
}
